// ParallelSTBlock_46806553592231
// MI455X (gfx1250) — compile-verified, matrix operand order checked by reading, not run on hardware
//
#include <hip/hip_runtime.h>
#include <math.h>

// ---------------- problem constants ----------------
constexpr int Bn   = 2;            // batch
constexpr int Nn   = 1024;         // nodes
constexpr int Tn   = 8;            // time steps
constexpr int Dn   = 128;          // model dim
constexpr int Hn   = 4;            // heads
constexpr int En   = 8192;         // edges (before self loops)
constexpr int ETOT = En + Nn;      // 9216 edges incl. self loops
constexpr int DHn  = Dn / Hn;      // 32
constexpr int Mtok = Bn * Nn * Tn; // 16384 tokens
constexpr int SEQ  = Bn * Nn;      // 2048 sequences
constexpr float NEG_SLOPE = 0.2f;

typedef __attribute__((ext_vector_type(16))) _Float16 v16h;
typedef __attribute__((ext_vector_type(8)))  _Float16 h8;
typedef __attribute__((ext_vector_type(8)))  float    v8f;

// ---------------- helpers ----------------
__device__ inline float wave_sum32(float v) {
#pragma unroll
  for (int m = 16; m >= 1; m >>= 1) v += __shfl_xor(v, m, 32);
  return v;
}

// monotonic float->uint key for atomic max
__device__ inline unsigned fkey(float x) {
  unsigned u = __float_as_uint(x);
  return (u & 0x80000000u) ? ~u : (u | 0x80000000u);
}
__device__ inline float unfkey(unsigned k) {
  unsigned u = (k & 0x80000000u) ? (k & 0x7fffffffu) : ~k;
  return __uint_as_float(u);
}

// ---------------- init / convert ----------------
__global__ void zero_f32_kernel(float* p, int n) {
  int i = blockIdx.x * blockDim.x + threadIdx.x;
  if (i < n) p[i] = 0.0f;
}

__global__ void f32_to_f16_kernel(const float* __restrict__ s, _Float16* __restrict__ d, int n) {
  int i = blockIdx.x * blockDim.x + threadIdx.x;
  if (i < n) d[i] = (_Float16)s[i];
}

// WT[n*K + k] = (f16) W[k*Nc + n]   (column-major f16 weights)
__global__ void wtrans_f16_kernel(const float* __restrict__ W, _Float16* __restrict__ WT,
                                  int K, int Nc) {
  int i = blockIdx.x * blockDim.x + threadIdx.x;
  if (i >= K * Nc) return;
  int k = i / Nc, n = i - k * Nc;
  WT[(size_t)n * K + k] = (_Float16)W[i];
}

// ---------------- generic WMMA GEMM (f16 in, f32 accum/out) ----------------
// Y[M,Nc] = act(X[M,K] @ WT^T + bias).  X row-major f16, WT col-major f16.
// One wave per block; wave computes a 16x64 strip (A frag reused over 4 N tiles).
// All fragment loads are contiguous 16-byte b128 loads. act: 0=none,1=GELU,2=sigmoid
__global__ __launch_bounds__(32)
void wmma_gemm_f16_kernel(const _Float16* __restrict__ X, const _Float16* __restrict__ WT,
                          const float* __restrict__ bias, float* __restrict__ Y,
                          int M, int K, int Nc, int act) {
  const int lane = threadIdx.x;
  const int m0   = blockIdx.x * 16;
  const int n0   = blockIdx.y * 64;
  const int half = lane >> 4;      // 0: K{0..7,16..23}  1: K{8..15,24..31}
  const int l16  = lane & 15;
  const int row  = m0 + l16;
  const int kb   = half * 8;

  v8f acc[4] = {};
  for (int k0 = 0; k0 < K; k0 += 32) {
    const _Float16* xp = X + (size_t)row * K + k0 + kb;
    h8 alo = *(const h8*)xp;
    h8 ahi = *(const h8*)(xp + 16);
    v16h a;
#pragma unroll
    for (int i = 0; i < 8; ++i) { a[i] = alo[i]; a[8 + i] = ahi[i]; }
#pragma unroll
    for (int j = 0; j < 4; ++j) {
      const int col = n0 + j * 16 + l16;
      const _Float16* wp = WT + (size_t)col * K + k0 + kb;
      h8 blo = *(const h8*)wp;
      h8 bhi = *(const h8*)(wp + 16);
      v16h b;
#pragma unroll
      for (int i = 0; i < 8; ++i) { b[i] = blo[i]; b[8 + i] = bhi[i]; }
      acc[j] = __builtin_amdgcn_wmma_f32_16x16x32_f16(
          false, a, false, b, (short)0, acc[j], false, false);
    }
  }
  // C/D layout: VGPR i -> row m0+i (lanes 0-15) / m0+i+8 (lanes 16-31), col = l16
#pragma unroll
  for (int j = 0; j < 4; ++j) {
    const int col = n0 + j * 16 + l16;
    const float bv = bias ? bias[col] : 0.0f;
#pragma unroll
    for (int i = 0; i < 8; ++i) {
      const int r = m0 + i + half * 8;
      float v = acc[j][i] + bv;
      if (act == 1)      v = 0.5f * v * (1.0f + erff(v * 0.70710678118f));
      else if (act == 2) v = 1.0f / (1.0f + __expf(-v));
      Y[(size_t)r * Nc + col] = v;
    }
  }
}

// ---------------- GAT edge pipeline ----------------
__global__ void edge_logits_kernel(const float* __restrict__ xl, const float* __restrict__ xr,
                                   const int* __restrict__ ei, const float* __restrict__ att,
                                   float* __restrict__ logits, unsigned* __restrict__ mx) {
  int idx = blockIdx.x * blockDim.x + threadIdx.x;
  const int total = ETOT * Bn * Tn * Hn;
  if (idx >= total) return;
  int h = idx & 3, t = (idx >> 2) & 7, b = (idx >> 5) & 1, e = idx >> 6;
  int sn = (e < En) ? ei[e]      : (e - En);
  int dn = (e < En) ? ei[En + e] : (e - En);
  const float* pl = xl + ((((size_t)b * Nn + sn) * Tn + t) * Hn + h) * Dn;
  const float* pr = xr + ((((size_t)b * Nn + dn) * Tn + t) * Hn + h) * Dn;
  const float* pa = att + h * Dn;
  float s = 0.0f;
  for (int d = 0; d < Dn; ++d) {
    float v = pl[d] + pr[d];
    v = (v > 0.0f) ? v : NEG_SLOPE * v;
    s += v * pa[d];
  }
  logits[idx] = s;
  atomicMax(&mx[(((size_t)dn * Bn + b) * Tn + t) * Hn + h], fkey(s));
}

__global__ void edge_expsum_kernel(float* __restrict__ logits, const unsigned* __restrict__ mx,
                                   const int* __restrict__ ei, float* __restrict__ ssum) {
  int idx = blockIdx.x * blockDim.x + threadIdx.x;
  const int total = ETOT * Bn * Tn * Hn;
  if (idx >= total) return;
  int h = idx & 3, t = (idx >> 2) & 7, b = (idx >> 5) & 1, e = idx >> 6;
  int dn = (e < En) ? ei[En + e] : (e - En);
  size_t si = (((size_t)dn * Bn + b) * Tn + t) * Hn + h;
  float p = __expf(logits[idx] - unfkey(mx[si]));
  logits[idx] = p;
  atomicAdd(&ssum[si], p);
}

__global__ void edge_scatter_kernel(const float* __restrict__ p, const float* __restrict__ ssum,
                                    const float* __restrict__ xl, const int* __restrict__ ei,
                                    float* __restrict__ agg) {
  long long idx = (long long)blockIdx.x * blockDim.x + threadIdx.x;
  const long long total = (long long)ETOT * Bn * Tn * Hn * Dn;
  if (idx >= total) return;
  int d = (int)(idx & 127);
  long long eidx = idx >> 7;                  // (e,b,t,h) flat index
  int h = (int)(eidx & 3), t = (int)((eidx >> 2) & 7), b = (int)((eidx >> 5) & 1);
  int e = (int)(eidx >> 6);
  int sn = (e < En) ? ei[e]      : (e - En);
  int dn = (e < En) ? ei[En + e] : (e - En);
  size_t si = (((size_t)dn * Bn + b) * Tn + t) * Hn + h;
  float alpha = p[eidx] / ssum[si];
  float ml = xl[((((size_t)b * Nn + sn) * Tn + t) * Hn + h) * Dn + d];
  atomicAdd(&agg[((((size_t)b * Nn + dn) * Tn + t) * Hn + h) * Dn + d], alpha * ml);
}

// x_sp = LN( mean_h(agg) + gat_b ; lns )  — one wave per token
__global__ __launch_bounds__(256)
void spatial_finalize_kernel(const float* __restrict__ agg, const float* __restrict__ gat_b,
                             const float* __restrict__ g, const float* __restrict__ bta,
                             float* __restrict__ xsp) {
  int lane = threadIdx.x & 31;
  int tok  = blockIdx.x * 8 + (threadIdx.x >> 5);
  if (tok >= Mtok) return;
  float v[4];
#pragma unroll
  for (int i = 0; i < 4; ++i) {
    int d = lane + 32 * i;
    float s = 0.0f;
#pragma unroll
    for (int h = 0; h < Hn; ++h) s += agg[((size_t)tok * Hn + h) * Dn + d];
    v[i] = s * (1.0f / Hn) + gat_b[d];
  }
  float m = wave_sum32(v[0] + v[1] + v[2] + v[3]) * (1.0f / Dn);
  float var = 0.0f;
#pragma unroll
  for (int i = 0; i < 4; ++i) { float c = v[i] - m; var += c * c; }
  var = wave_sum32(var) * (1.0f / Dn);
  float rs = rsqrtf(var + 1e-5f);
#pragma unroll
  for (int i = 0; i < 4; ++i) {
    int d = lane + 32 * i;
    xsp[(size_t)tok * Dn + d] = (v[i] - m) * rs * g[d] + bta[d];
  }
}

// ---------------- temporal attention (T=8, DH=32): one thread per (seq,h,t) ----------------
__global__ void attention_kernel(const float* __restrict__ qkv, float* __restrict__ o) {
  int idx = blockIdx.x * blockDim.x + threadIdx.x;
  if (idx >= SEQ * Hn * Tn) return;
  int t = idx & 7, h = (idx >> 3) & 3, seq = idx >> 5;
  const size_t base = (size_t)seq * Tn * (3 * Dn);
  const float* q = qkv + base + (size_t)t * (3 * Dn) + h * DHn;
  float qr[DHn];
#pragma unroll
  for (int d = 0; d < DHn; ++d) qr[d] = q[d];
  float sc[Tn];
  float mxv = -1e30f;
#pragma unroll
  for (int s = 0; s < Tn; ++s) {
    const float* k = qkv + base + (size_t)s * (3 * Dn) + Dn + h * DHn;
    float acc = 0.0f;
#pragma unroll
    for (int d = 0; d < DHn; ++d) acc += qr[d] * k[d];
    sc[s] = acc * 0.17677669529f;  // 1/sqrt(32)
    mxv = fmaxf(mxv, sc[s]);
  }
  float den = 0.0f;
#pragma unroll
  for (int s = 0; s < Tn; ++s) { sc[s] = __expf(sc[s] - mxv); den += sc[s]; }
  float inv = 1.0f / den;
  float ov[DHn];
#pragma unroll
  for (int d = 0; d < DHn; ++d) ov[d] = 0.0f;
#pragma unroll
  for (int s = 0; s < Tn; ++s) {
    const float* v = qkv + base + (size_t)s * (3 * Dn) + 2 * Dn + h * DHn;
    float w = sc[s] * inv;
#pragma unroll
    for (int d = 0; d < DHn; ++d) ov[d] += w * v[d];
  }
  float* op = o + ((size_t)seq * Tn + t) * Dn + h * DHn;
#pragma unroll
  for (int d = 0; d < DHn; ++d) op[d] = ov[d];
}

// ---------------- add + LayerNorm (one wave per token) ----------------
__global__ __launch_bounds__(256)
void add_ln_kernel(const float* __restrict__ a, const float* __restrict__ b,
                   const float* __restrict__ g, const float* __restrict__ bta,
                   float* __restrict__ y) {
  int lane = threadIdx.x & 31;
  int tok  = blockIdx.x * 8 + (threadIdx.x >> 5);
  if (tok >= Mtok) return;
  float v[4];
#pragma unroll
  for (int i = 0; i < 4; ++i) {
    int d = lane + 32 * i;
    v[i] = a[(size_t)tok * Dn + d] + b[(size_t)tok * Dn + d];
  }
  float m = wave_sum32(v[0] + v[1] + v[2] + v[3]) * (1.0f / Dn);
  float var = 0.0f;
#pragma unroll
  for (int i = 0; i < 4; ++i) { float c = v[i] - m; var += c * c; }
  var = wave_sum32(var) * (1.0f / Dn);
  float rs = rsqrtf(var + 1e-5f);
#pragma unroll
  for (int i = 0; i < 4; ++i) {
    int d = lane + 32 * i;
    y[(size_t)tok * Dn + d] = (v[i] - m) * rs * g[d] + bta[d];
  }
}

// ---------------- concat [x_sp | x_tp] straight into f16 ----------------
__global__ void concat_f16_kernel(const float* __restrict__ xsp, const float* __restrict__ xtp,
                                  _Float16* __restrict__ cat) {
  int idx = blockIdx.x * blockDim.x + threadIdx.x;
  if (idx >= Mtok * 2 * Dn) return;
  int c = idx & 255, tok = idx >> 8;
  float v = (c < Dn) ? xsp[(size_t)tok * Dn + c] : xtp[(size_t)tok * Dn + (c - Dn)];
  cat[idx] = (_Float16)v;
}

// ---------------- gated fusion + final LN ----------------
__global__ __launch_bounds__(256)
void fuse_ln_kernel(const float* __restrict__ gate, const float* __restrict__ xsp,
                    const float* __restrict__ xtp, const float* __restrict__ x,
                    const float* __restrict__ g, const float* __restrict__ bta,
                    float* __restrict__ out) {
  int lane = threadIdx.x & 31;
  int tok  = blockIdx.x * 8 + (threadIdx.x >> 5);
  if (tok >= Mtok) return;
  float v[4];
#pragma unroll
  for (int i = 0; i < 4; ++i) {
    int d = lane + 32 * i;
    size_t o = (size_t)tok * Dn + d;
    float gv = gate[o];
    v[i] = gv * xsp[o] + (1.0f - gv) * xtp[o] + x[o];
  }
  float m = wave_sum32(v[0] + v[1] + v[2] + v[3]) * (1.0f / Dn);
  float var = 0.0f;
#pragma unroll
  for (int i = 0; i < 4; ++i) { float c = v[i] - m; var += c * c; }
  var = wave_sum32(var) * (1.0f / Dn);
  float rs = rsqrtf(var + 1e-5f);
#pragma unroll
  for (int i = 0; i < 4; ++i) {
    int d = lane + 32 * i;
    out[(size_t)tok * Dn + d] = (v[i] - m) * rs * g[d] + bta[d];
  }
}

// ---------------- launch ----------------
extern "C" void kernel_launch(void* const* d_in, const int* in_sizes, int n_in,
                              void* d_out, int out_size, void* d_ws, size_t ws_size,
                              hipStream_t stream) {
  const float* x       = (const float*)d_in[0];
  const int*   ei      = (const int*)  d_in[1];
  const float* gat_wl  = (const float*)d_in[2];
  const float* gat_wr  = (const float*)d_in[3];
  const float* gat_att = (const float*)d_in[4];
  const float* gat_b   = (const float*)d_in[5];
  const float* in_w    = (const float*)d_in[6];
  const float* in_b    = (const float*)d_in[7];
  const float* out_w   = (const float*)d_in[8];
  const float* out_b   = (const float*)d_in[9];
  const float* ffn_w1  = (const float*)d_in[10];
  const float* ffn_b1  = (const float*)d_in[11];
  const float* ffn_w2  = (const float*)d_in[12];
  const float* ffn_b2  = (const float*)d_in[13];
  const float* fus_w   = (const float*)d_in[14];
  const float* fus_b   = (const float*)d_in[15];
  const float* lns_g   = (const float*)d_in[16];
  const float* lns_b   = (const float*)d_in[17];
  const float* lnt1_g  = (const float*)d_in[18];
  const float* lnt1_b  = (const float*)d_in[19];
  const float* lnt2_g  = (const float*)d_in[20];
  const float* lnt2_b  = (const float*)d_in[21];
  const float* lnf_g   = (const float*)d_in[22];
  const float* lnf_b   = (const float*)d_in[23];
  float* out = (float*)d_out;

  // ---- workspace layout (float units); big slabs reused across phases ----
  float* ws = (float*)d_ws;
  const size_t SLAB = (size_t)Bn * Nn * Tn * Hn * Dn;   // 8,388,608
  const size_t TD   = (size_t)Mtok * Dn;                // 2,097,152
  float* XL     = ws;                    // xl  -> later QKV
  float* XR     = ws + SLAB;             // xr  -> later O / X1 / FFNO / G
  float* AGG    = ws + 2 * SLAB;         // agg -> later FFNH
  float* LOGITS = ws + 3 * SLAB;                        // 589,824
  unsigned* MX  = (unsigned*)(LOGITS + (size_t)ETOT * Bn * Tn * Hn); // 65,536
  float* SSUM   = (float*)MX + (size_t)Nn * Bn * Tn * Hn;            // 65,536
  float* XSP    = SSUM + (size_t)Nn * Bn * Tn * Hn;     // persists
  float* XTP    = XSP + TD;                             // persists
  float* OPRJ   = XTP + TD;                             // f32 out-proj result
  // f16 staging area
  _Float16* XH   = (_Float16*)(OPRJ + TD);      // x in f16         (2,097,152 h)
  _Float16* AH   = XH + TD;                     // activation f16   (8,388,608 h) reused
  _Float16* CATH = AH + (size_t)Mtok * 512;     // concat f16       (4,194,304 h)
  _Float16* WT   = CATH + (size_t)Mtok * 256;   // transposed weights
  _Float16* WT_GL  = WT;                        // 128x512 -> 65,536
  _Float16* WT_GR  = WT_GL + 65536;
  _Float16* WT_IN  = WT_GR + 65536;             // 128x384 -> 49,152
  _Float16* WT_OUT = WT_IN + 49152;             // 128x128 -> 16,384
  _Float16* WT_F1  = WT_OUT + 16384;            // 128x512 -> 65,536
  _Float16* WT_F2  = WT_F1 + 65536;             // 512x128 -> 65,536
  _Float16* WT_FU  = WT_F2 + 65536;             // 256x128 -> 32,768
  float* QKV  = XL;
  float* O    = XR;
  float* X1   = XR + TD;
  float* FFNO = XR + 2 * TD;
  float* G    = XR + 3 * TD;
  float* FFNH = AGG;

  // zero agg + logits + mx + ssum (contiguous region)
  {
    int n = (int)(SLAB + (size_t)ETOT * Bn * Tn * Hn + 2 * (size_t)Nn * Bn * Tn * Hn);
    zero_f32_kernel<<<(n + 255) / 256, 256, 0, stream>>>(AGG, n);
  }

  // ---- weight transposes + x conversion (tiny) ----
  wtrans_f16_kernel<<<(128 * 512 + 255) / 256, 256, 0, stream>>>(gat_wl, WT_GL, 128, 512);
  wtrans_f16_kernel<<<(128 * 512 + 255) / 256, 256, 0, stream>>>(gat_wr, WT_GR, 128, 512);
  wtrans_f16_kernel<<<(128 * 384 + 255) / 256, 256, 0, stream>>>(in_w,   WT_IN, 128, 384);
  wtrans_f16_kernel<<<(128 * 128 + 255) / 256, 256, 0, stream>>>(out_w,  WT_OUT, 128, 128);
  wtrans_f16_kernel<<<(128 * 512 + 255) / 256, 256, 0, stream>>>(ffn_w1, WT_F1, 128, 512);
  wtrans_f16_kernel<<<(512 * 128 + 255) / 256, 256, 0, stream>>>(ffn_w2, WT_F2, 512, 128);
  wtrans_f16_kernel<<<(256 * 128 + 255) / 256, 256, 0, stream>>>(fus_w,  WT_FU, 256, 128);
  f32_to_f16_kernel<<<(int)((TD + 255) / 256), 256, 0, stream>>>(x, XH, (int)TD);

  // ---- spatial (GAT) branch ----
  wmma_gemm_f16_kernel<<<dim3(Mtok / 16, 512 / 64), 32, 0, stream>>>(XH, WT_GL, nullptr, XL, Mtok, Dn, 512, 0);
  wmma_gemm_f16_kernel<<<dim3(Mtok / 16, 512 / 64), 32, 0, stream>>>(XH, WT_GR, nullptr, XR, Mtok, Dn, 512, 0);
  {
    int n = ETOT * Bn * Tn * Hn;
    edge_logits_kernel<<<(n + 255) / 256, 256, 0, stream>>>(XL, XR, ei, gat_att, LOGITS, MX);
    edge_expsum_kernel<<<(n + 255) / 256, 256, 0, stream>>>(LOGITS, MX, ei, SSUM);
    long long ns = (long long)n * Dn;
    edge_scatter_kernel<<<(unsigned)((ns + 255) / 256), 256, 0, stream>>>(LOGITS, SSUM, XL, ei, AGG);
  }
  spatial_finalize_kernel<<<Mtok / 8, 256, 0, stream>>>(AGG, gat_b, lns_g, lns_b, XSP);

  // ---- temporal branch ----
  wmma_gemm_f16_kernel<<<dim3(Mtok / 16, 384 / 64), 32, 0, stream>>>(XH, WT_IN, in_b, QKV, Mtok, Dn, 384, 0);
  attention_kernel<<<(SEQ * Hn * Tn + 255) / 256, 256, 0, stream>>>(QKV, O);
  f32_to_f16_kernel<<<(int)((TD + 255) / 256), 256, 0, stream>>>(O, AH, (int)TD);
  wmma_gemm_f16_kernel<<<dim3(Mtok / 16, 128 / 64), 32, 0, stream>>>(AH, WT_OUT, out_b, OPRJ, Mtok, Dn, 128, 0);
  add_ln_kernel<<<Mtok / 8, 256, 0, stream>>>(x, OPRJ, lnt1_g, lnt1_b, X1);
  f32_to_f16_kernel<<<(int)((TD + 255) / 256), 256, 0, stream>>>(X1, AH, (int)TD);
  wmma_gemm_f16_kernel<<<dim3(Mtok / 16, 512 / 64), 32, 0, stream>>>(AH, WT_F1, ffn_b1, FFNH, Mtok, Dn, 512, 1);
  f32_to_f16_kernel<<<(Mtok * 512 + 255) / 256, 256, 0, stream>>>(FFNH, AH, Mtok * 512);
  wmma_gemm_f16_kernel<<<dim3(Mtok / 16, 128 / 64), 32, 0, stream>>>(AH, WT_F2, ffn_b2, FFNO, Mtok, 512, 128, 0);
  add_ln_kernel<<<Mtok / 8, 256, 0, stream>>>(X1, FFNO, lnt2_g, lnt2_b, XTP);

  // ---- fusion ----
  concat_f16_kernel<<<(Mtok * 2 * Dn + 255) / 256, 256, 0, stream>>>(XSP, XTP, CATH);
  wmma_gemm_f16_kernel<<<dim3(Mtok / 16, 128 / 64), 32, 0, stream>>>(CATH, WT_FU, fus_b, G, Mtok, 2 * Dn, 128, 2);
  fuse_ln_kernel<<<Mtok / 8, 256, 0, stream>>>(G, XSP, XTP, x, lnf_g, lnf_b, out);
}
